// BayesianDropoutLayer_83966610637560
// MI455X (gfx1250) — matrix operand order repaired and checked by reading, not compiled, not run on hardware
//
#include <hip/hip_runtime.h>

// ---------------------------------------------------------------------------
// Y[8192,2048] = X[8192,2048] @ (mask[k] * M[k,2048]) + m   (f32 reference)
// bf16x3 split-precision GEMM on CDNA5 WMMA (v_wmma_f32_16x16x32_bf16).
// Workgroup: 512 threads = 16 waves (4x4), tile 128x128, K-chunk 32.
// ---------------------------------------------------------------------------

typedef __attribute__((ext_vector_type(16))) __bf16 v16bf;
typedef __attribute__((ext_vector_type(8)))  float  v8f;
typedef __attribute__((ext_vector_type(4)))  float  f4;

#define BATCH 8192
#define KDIM  2048
#define UNITS 2048

#define BM 128   // workgroup tile rows (batch)
#define BN 128   // workgroup tile cols (units)
#define BK 32    // K chunk == bf16 WMMA K
#define LDA 48   // padded LDS stride (ushort units): 96B rows, 16B-aligned, bank-staggered
#define LDB 48

union FragB16 { v16bf v; uint4 q[2]; };

// Error-free two-term bf16 split: hi = truncate(x), lo = RNE(x - hi).
__device__ __forceinline__ void split_bf16(float x, unsigned short& h, unsigned short& l) {
    unsigned u = __float_as_uint(x);
    h = (unsigned short)(u >> 16);                 // truncated bf16 (exact residual)
    float hf = __uint_as_float(u & 0xFFFF0000u);
    __bf16 lb = (__bf16)(x - hf);                  // RNE convert of residual
    union { __bf16 b; unsigned short s; } cv;
    cv.b = lb;
    l = cv.s;
}

__global__ __launch_bounds__(512)
void masked_gemm_bf16x3_kernel(const float* __restrict__ X,
                               const float* __restrict__ M,
                               const float* __restrict__ bias,
                               const int*   __restrict__ mask,
                               float* __restrict__ Out)
{
    __shared__ unsigned short Ah[BM][LDA];   // X tile, hi plane, [m][k]
    __shared__ unsigned short Al[BM][LDA];   // X tile, lo plane
    __shared__ unsigned short Bh[BN][LDB];   // W tile (mask*M), hi plane, [n][k] (transposed)
    __shared__ unsigned short Bl[BN][LDB];   // W tile, lo plane

    const int tid  = threadIdx.x;
    const int lane = tid & 31;
    const int wave = tid >> 5;               // 0..15

    const int blockM = blockIdx.y * BM;
    const int blockN = blockIdx.x * BN;

    // 16 waves arranged 4(M) x 4(N); each wave owns a 32x32 patch.
    const int waveM = (wave & 3) * 32;
    const int waveN = (wave >> 2) * 32;

    // Cooperative global-load mapping (512 threads).
    const int ax_r = tid >> 3;          // 0..63  (X tile row within 64-row pass)
    const int ax_c = (tid & 7) * 4;     // 0..28  (k offset, float4 granularity)
    const int bw_r = tid >> 5;          // 0..15  (W tile k-row within 16-row pass)
    const int bw_c = (tid & 31) * 4;    // 0..124 (n offset, float4 granularity)

    v8f acc[2][2];
#pragma unroll
    for (int i = 0; i < 2; ++i)
#pragma unroll
        for (int j = 0; j < 2; ++j)
            acc[i][j] = (v8f)0.0f;

    f4    xReg[2];
    f4    wReg[2];
    float wScale[2];

    const int row = lane & 15;          // row/col within a 16x16 fragment
    const int sel = lane >> 4;          // half-wave select

    // ---- prologue: load K-chunk 0 into registers -------------------------
    {
#pragma unroll
        for (int p = 0; p < 2; ++p)
            xReg[p] = *(const f4*)(X + (size_t)(blockM + ax_r + 64 * p) * KDIM + ax_c);
#pragma unroll
        for (int p = 0; p < 2; ++p) {
            const int kr = bw_r + 16 * p;
            wReg[p]   = *(const f4*)(M + (size_t)kr * UNITS + blockN + bw_c);
            wScale[p] = (float)mask[kr];
        }
    }

    for (int kc = 0; kc < KDIM; kc += BK) {
        __syncthreads();   // previous chunk's compute finished reading LDS

        // ---- stage registers -> LDS (split into hi/lo bf16 planes) -------
#pragma unroll
        for (int p = 0; p < 2; ++p) {
            const int r = ax_r + 64 * p;
#pragma unroll
            for (int j = 0; j < 4; ++j) {
                unsigned short h, l;
                split_bf16(xReg[p][j], h, l);
                Ah[r][ax_c + j] = h;
                Al[r][ax_c + j] = l;
            }
        }
#pragma unroll
        for (int p = 0; p < 2; ++p) {
            const int k = bw_r + 16 * p;
#pragma unroll
            for (int j = 0; j < 4; ++j) {
                unsigned short h, l;
                split_bf16(wScale[p] * wReg[p][j], h, l);
                Bh[bw_c + j][k] = h;                 // transposed store: [n][k]
                Bl[bw_c + j][k] = l;
            }
        }

        __syncthreads();   // tile visible to all waves

        // ---- issue next chunk's global loads (overlap with WMMA below) ---
        if (kc + BK < KDIM) {
            const int kn = kc + BK;
#pragma unroll
            for (int p = 0; p < 2; ++p)
                xReg[p] = *(const f4*)(X + (size_t)(blockM + ax_r + 64 * p) * KDIM + kn + ax_c);
#pragma unroll
            for (int p = 0; p < 2; ++p) {
                const int kr = kn + bw_r + 16 * p;
                wReg[p]   = *(const f4*)(M + (size_t)kr * UNITS + blockN + bw_c);
                wScale[p] = (float)mask[kr];
            }
        }

        // ---- gather WMMA fragments from LDS ------------------------------
        // A (16-bit, 16x32): lane<16 -> row, K{0..7,16..23}; lane>=16 -> K{8..15,24..31}
        FragB16 a[2][2];   // [ti][0=hi,1=lo]
#pragma unroll
        for (int ti = 0; ti < 2; ++ti) {
            const unsigned short* bh = &Ah[waveM + ti * 16 + row][0];
            const unsigned short* bl = &Al[waveM + ti * 16 + row][0];
            a[ti][0].q[0] = *(const uint4*)(bh + sel * 8);
            a[ti][0].q[1] = *(const uint4*)(bh + 16 + sel * 8);
            a[ti][1].q[0] = *(const uint4*)(bl + sel * 8);
            a[ti][1].q[1] = *(const uint4*)(bl + 16 + sel * 8);
        }
        // B (16-bit, 32x16): lane<16 -> col, K{0..7,16..23}; lane>=16 -> K{8..15,24..31} (matches A)
        FragB16 b[2][2];
#pragma unroll
        for (int tj = 0; tj < 2; ++tj) {
            const unsigned short* bh = &Bh[waveN + tj * 16 + row][0];
            const unsigned short* bl = &Bl[waveN + tj * 16 + row][0];
            b[tj][0].q[0] = *(const uint4*)(bh + sel * 8);
            b[tj][0].q[1] = *(const uint4*)(bh + 16 + sel * 8);
            b[tj][1].q[0] = *(const uint4*)(bl + sel * 8);
            b[tj][1].q[1] = *(const uint4*)(bl + 16 + sel * 8);
        }

        // ---- 12 WMMAs: hi*hi + hi*lo + lo*hi (lo*lo dropped) -------------
#pragma unroll
        for (int ti = 0; ti < 2; ++ti) {
#pragma unroll
            for (int tj = 0; tj < 2; ++tj) {
                acc[ti][tj] = __builtin_amdgcn_wmma_f32_16x16x32_bf16(
                    false, a[ti][0].v, false, b[tj][0].v, (short)0, acc[ti][tj], false, false);
                acc[ti][tj] = __builtin_amdgcn_wmma_f32_16x16x32_bf16(
                    false, a[ti][0].v, false, b[tj][1].v, (short)0, acc[ti][tj], false, false);
                acc[ti][tj] = __builtin_amdgcn_wmma_f32_16x16x32_bf16(
                    false, a[ti][1].v, false, b[tj][0].v, (short)0, acc[ti][tj], false, false);
            }
        }
    }

    // ---- epilogue: C layout VGPR i -> M = i (+8 for upper half-wave) -----
    // Output is write-once (64 MB): stream it past L2 with non-temporal stores
    // so L2 keeps holding X/M panels for the other N-blocks.
#pragma unroll
    for (int ti = 0; ti < 2; ++ti) {
#pragma unroll
        for (int tj = 0; tj < 2; ++tj) {
            const int r0 = blockM + waveM + ti * 16 + sel * 8;
            const int c  = blockN + waveN + tj * 16 + row;
            const float bv = bias[c];
#pragma unroll
            for (int i = 0; i < 8; ++i)
                __builtin_nontemporal_store(acc[ti][tj][i] + bv,
                                            &Out[(size_t)(r0 + i) * UNITS + c]);
        }
    }
}

extern "C" void kernel_launch(void* const* d_in, const int* in_sizes, int n_in,
                              void* d_out, int out_size, void* d_ws, size_t ws_size,
                              hipStream_t stream) {
    (void)in_sizes; (void)n_in; (void)out_size; (void)d_ws; (void)ws_size;
    const float* X    = (const float*)d_in[0];
    const float* M    = (const float*)d_in[1];
    const float* m    = (const float*)d_in[2];
    const int*   mask = (const int*)  d_in[3];
    float*       out  = (float*)d_out;

    dim3 grid(UNITS / BN, BATCH / BM);   // 16 x 64 workgroups
    masked_gemm_bf16x3_kernel<<<grid, 512, 0, stream>>>(X, M, m, mask, out);
}
